// guided_CAM_Module_36730560315896
// MI455X (gfx1250) — matrix operand order repaired and checked by reading, not compiled, not run on hardware
//
#include <hip/hip_runtime.h>
#include <cstdint>
#include <cstddef>

// Problem dims (from reference)
#define B_  4
#define C_  2560
#define Q_  512
#define O_  512
#define N_  9216   // 96*96
#define BN_EPS_ 1e-5f

typedef __attribute__((ext_vector_type(16))) __bf16 v16bf;
typedef __attribute__((ext_vector_type(8)))  __bf16 v8bf;
typedef __attribute__((ext_vector_type(4)))  __bf16 v4bf;
typedef __attribute__((ext_vector_type(8)))  float  v8f;

// ---------------------------------------------------------------------------
// CDNA5 async global->LDS copy (ASYNCcnt-tracked), CDNA5 ISA
// Per-lane: 16 bytes from global address vaddr to LDS offset vdst.
// ---------------------------------------------------------------------------
__device__ inline void async_copy_b128(const __bf16* g, __bf16* l) {
  unsigned lds_addr = (unsigned)(unsigned long long)(void*)l;  // low 32 bits = LDS offset
  unsigned long long gaddr = (unsigned long long)(const void*)g;
  asm volatile("global_load_async_to_lds_b128 %0, %1, off"
               :: "v"(lds_addr), "v"(gaddr)
               : "memory");
}
__device__ inline void wait_async0() {
  asm volatile("s_wait_asynccnt 0x0" ::: "memory");
}

// ---------------------------------------------------------------------------
// Fragment loaders matching CDNA5 WMMA VGPR layouts (CDNA5 ISA).
// Used against LDS tiles (stride = 32 elements). Source is row-major with the
// dot dimension contiguous.
// A (16x32, MxK): lane L -> M=L%16; elems 0..7 = K (L>=16?8:0)+0..7,
//                 elems 8..15 = K 16+(L>=16?8:0)+0..7.
// ---------------------------------------------------------------------------
__device__ inline v16bf load_a_frag(const __bf16* src, int row0, int ld, int lane) {
  int r  = lane & 15;
  int hi = lane >> 4;
  const __bf16* p = src + (row0 + r) * ld + hi * 8;
  v8bf lo = *(const v8bf*)(p);
  v8bf hc = *(const v8bf*)(p + 16);
  v16bf a;
#pragma unroll
  for (int i = 0; i < 8; ++i) { a[i] = lo[i]; a[i + 8] = hc[i]; }
  return a;
}

// B (32x16, KxN): lane L -> col=L%16; elems 0..7 = K (L>=16?8:0)+0..7,
// elems 8..15 = K 16+(L>=16?8:0)+0..7 (same order as A).
__device__ inline v16bf load_b_frag(const __bf16* srcT, int col0, int ld, int lane) {
  int c  = lane & 15;
  int hi = lane >> 4;
  const __bf16* p = srcT + (col0 + c) * ld + hi * 8;
  v8bf lo = *(const v8bf*)(p);
  v8bf hc = *(const v8bf*)(p + 16);
  v16bf b;
#pragma unroll
  for (int i = 0; i < 8; ++i) { b[i] = lo[i]; b[i + 8] = hc[i]; }
  return b;
}

// ---------------------------------------------------------------------------
// Issue async copies for one 128x32 A tile and one 128x32 B tile (bf16).
// 128 threads * 4 chunks * 16B = 8KB per tile.
// ---------------------------------------------------------------------------
__device__ inline void issue_tile_loads(const __bf16* A, const __bf16* Bt,
                                        size_t lda, size_t ldb,
                                        size_t m0, size_t n0, size_t k0,
                                        __bf16* la, __bf16* lb, int tid) {
#pragma unroll
  for (int j = 0; j < 4; ++j) {
    int chunk = tid + 128 * j;   // 0..511
    int row = chunk >> 2;
    int sub = chunk & 3;
    async_copy_b128(A  + (m0 + row) * lda + k0 + sub * 8, la + row * 32 + sub * 8);
    async_copy_b128(Bt + (n0 + row) * ldb + k0 + sub * 8, lb + row * 32 + sub * 8);
  }
}

// Per-wave 64x64 compute step from the 128x32 LDS tiles.
__device__ inline void compute_step(const __bf16* la, const __bf16* lb,
                                    int wm, int wn, int lane, v8f acc[4][4]) {
  v16bf a[4], b[4];
#pragma unroll
  for (int mi = 0; mi < 4; ++mi) a[mi] = load_a_frag(la, wm * 64 + mi * 16, 32, lane);
#pragma unroll
  for (int ni = 0; ni < 4; ++ni) b[ni] = load_b_frag(lb, wn * 64 + ni * 16, 32, lane);
#pragma unroll
  for (int mi = 0; mi < 4; ++mi)
#pragma unroll
    for (int ni = 0; ni < 4; ++ni)
      acc[mi][ni] = __builtin_amdgcn_wmma_f32_16x16x32_bf16(
          false, a[mi], false, b[ni], (short)0, acc[mi][ni], false, false);
}

// Shared double-buffered K-loop. 4 waves (128 thr) per block, 128x128 tile.
#define GEMM_CORE(A_, Bt_, LDA_, LDB_, KDIM_)                                  \
  __shared__ __attribute__((aligned(64))) __bf16 lsA[2][128 * 32];             \
  __shared__ __attribute__((aligned(64))) __bf16 lsB[2][128 * 32];             \
  int tid = threadIdx.x;                                                       \
  int lane = tid & 31;                                                         \
  int w = tid >> 5, wm = w >> 1, wn = w & 1;                                   \
  size_t m0 = (size_t)blockIdx.y * 128;                                        \
  size_t n0 = (size_t)blockIdx.x * 128;                                        \
  v8f zero = {0.f, 0.f, 0.f, 0.f, 0.f, 0.f, 0.f, 0.f};                         \
  v8f acc[4][4];                                                               \
  _Pragma("unroll") for (int mi = 0; mi < 4; ++mi)                             \
      _Pragma("unroll") for (int ni = 0; ni < 4; ++ni) acc[mi][ni] = zero;     \
  const int nk = (KDIM_) / 32;                                                 \
  issue_tile_loads(A_, Bt_, LDA_, LDB_, m0, n0, 0, lsA[0], lsB[0], tid);       \
  for (int i = 0; i < nk; ++i) {                                               \
    wait_async0();        /* my copies into buf i&1 done            */         \
    __syncthreads();      /* everyone's copies done; prev compute done */      \
    if (i + 1 < nk)                                                            \
      issue_tile_loads(A_, Bt_, LDA_, LDB_, m0, n0, (size_t)(i + 1) * 32,      \
                       lsA[(i + 1) & 1], lsB[(i + 1) & 1], tid);               \
    compute_step(lsA[i & 1], lsB[i & 1], wm, wn, lane, acc);                   \
  }                                                                            \
  size_t m0w = m0 + (size_t)wm * 64;                                           \
  size_t n0w = n0 + (size_t)wn * 64;                                           \
  int col = lane & 15, hi = lane >> 4;

// ---------------------------------------------------------------------------
// fp32 -> bf16 flat convert (vector of 4)
// ---------------------------------------------------------------------------
__global__ void cvt_f32_to_bf16(const float* __restrict__ in,
                                __bf16* __restrict__ out, size_t n4) {
  size_t i = (size_t)blockIdx.x * blockDim.x + threadIdx.x;
  size_t stride = (size_t)gridDim.x * blockDim.x;
  for (; i < n4; i += stride) {
    float4 v = ((const float4*)in)[i];
    v4bf o;
    o[0] = (__bf16)v.x; o[1] = (__bf16)v.y; o[2] = (__bf16)v.z; o[3] = (__bf16)v.w;
    ((v4bf*)out)[i] = o;
  }
}

// ---------------------------------------------------------------------------
// Transpose + convert: x[b][C][N] fp32 -> xT[b][N][C] bf16 (32x32 LDS tiles)
// ---------------------------------------------------------------------------
__global__ void transpose_cvt(const float* __restrict__ x, __bf16* __restrict__ xT) {
  __shared__ __bf16 tile[32][33];
  size_t bz = blockIdx.z;
  size_t n0 = (size_t)blockIdx.x * 32;
  size_t c0 = (size_t)blockIdx.y * 32;
  const float* X  = x  + bz * (size_t)C_ * N_;
  __bf16*      XT = xT + bz * (size_t)N_ * C_;
  int tx = threadIdx.x, ty = threadIdx.y; // 32 x 8
#pragma unroll
  for (int j = 0; j < 4; ++j)
    tile[ty + j * 8][tx] = (__bf16)X[(c0 + ty + j * 8) * (size_t)N_ + n0 + tx];
  __syncthreads();
#pragma unroll
  for (int j = 0; j < 4; ++j)
    XT[(n0 + ty + j * 8) * (size_t)C_ + c0 + tx] = tile[tx][ty + j * 8];
}

// ---------------------------------------------------------------------------
// GEMM1: energy[b][q][c] = sum_n qb[b][q][n] * xb[b][c][n]   (K = 9216)
// ---------------------------------------------------------------------------
__global__ __launch_bounds__(128) void gemm_energy(const __bf16* __restrict__ qb,
                                                   const __bf16* __restrict__ xb,
                                                   float* __restrict__ energy) {
  size_t bz = blockIdx.z;
  const __bf16* A  = qb + bz * (size_t)Q_ * N_;
  const __bf16* Bt = xb + bz * (size_t)C_ * N_;
  GEMM_CORE(A, Bt, (size_t)N_, (size_t)N_, N_)

  float* E = energy + bz * (size_t)Q_ * C_;
#pragma unroll
  for (int mi = 0; mi < 4; ++mi)
#pragma unroll
    for (int ni = 0; ni < 4; ++ni)
#pragma unroll
      for (int r = 0; r < 8; ++r)
        E[(m0w + mi * 16 + r + hi * 8) * (size_t)C_ + n0w + ni * 16 + col] =
            acc[mi][ni][r];
}

// ---------------------------------------------------------------------------
// Row softmax: att[row][c] = exp(minE - e[c]) / sum_c exp(minE - e[c])
// (== softmax(max(e) - e) with the stability shift folded in)
// ---------------------------------------------------------------------------
__global__ void softmax_rows(const float* __restrict__ energy, __bf16* __restrict__ att) {
  __shared__ float red[256];
  size_t row = blockIdx.x; // b*Q + q
  const float* e = energy + row * (size_t)C_;
  int tid = threadIdx.x;

  float mn = 3.4e38f;
  for (int c = tid; c < C_; c += 256) mn = fminf(mn, e[c]);
  red[tid] = mn;
  __syncthreads();
  for (int s = 128; s > 0; s >>= 1) {
    if (tid < s) red[tid] = fminf(red[tid], red[tid + s]);
    __syncthreads();
  }
  mn = red[0];
  __syncthreads();

  float sum = 0.f;
  for (int c = tid; c < C_; c += 256) sum += __expf(mn - e[c]);
  red[tid] = sum;
  __syncthreads();
  for (int s = 128; s > 0; s >>= 1) {
    if (tid < s) red[tid] += red[tid + s];
    __syncthreads();
  }
  float inv = 1.0f / red[0];

  __bf16* a = att + row * (size_t)C_;
  for (int c = tid; c < C_; c += 256) a[c] = (__bf16)(__expf(mn - e[c]) * inv);
}

// ---------------------------------------------------------------------------
// GEMM2: out_c[b][q][n] = sum_c att[b][q][c] * x[b][c][n]   (K = 2560)
// epilogue: v = gamma*out_c + query ; store TRANSPOSED bf16 outcT[b][n][q]
// ---------------------------------------------------------------------------
__global__ __launch_bounds__(128) void gemm_attn(const __bf16* __restrict__ att,
                                                 const __bf16* __restrict__ xTb,
                                                 const float* __restrict__ query,
                                                 const float* __restrict__ gamma,
                                                 __bf16* __restrict__ outcT) {
  size_t bz = blockIdx.z;
  const __bf16* A  = att + bz * (size_t)Q_ * C_;
  const __bf16* Bt = xTb + bz * (size_t)N_ * C_;
  GEMM_CORE(A, Bt, (size_t)C_, (size_t)C_, C_)

  float g = gamma[0];
  const float* qry = query + bz * (size_t)Q_ * N_;
  __bf16* oT = outcT + bz * (size_t)N_ * Q_;
#pragma unroll
  for (int mi = 0; mi < 4; ++mi)
#pragma unroll
    for (int ni = 0; ni < 4; ++ni)
#pragma unroll
      for (int r = 0; r < 8; ++r) {
        size_t qrow = m0w + mi * 16 + r + hi * 8;
        size_t ncol = n0w + ni * 16 + col;
        float v = g * acc[mi][ni][r] + qry[qrow * (size_t)N_ + ncol];
        oT[ncol * (size_t)Q_ + qrow] = (__bf16)v;
      }
}

// ---------------------------------------------------------------------------
// GEMM3: y[b][o][n] = sum_q w[o][q] * out_c[b][q][n]   (K = 512)
// epilogue: BatchNorm (inference) + ReLU, fp32 store to d_out
// ---------------------------------------------------------------------------
__global__ __launch_bounds__(128) void gemm_conv_bn(const __bf16* __restrict__ wb,
                                                    const __bf16* __restrict__ outcT,
                                                    const float* __restrict__ bn_scale,
                                                    const float* __restrict__ bn_bias,
                                                    const float* __restrict__ bn_mean,
                                                    const float* __restrict__ bn_var,
                                                    float* __restrict__ out) {
  size_t bz = blockIdx.z;
  const __bf16* A  = wb;
  const __bf16* Bt = outcT + bz * (size_t)N_ * Q_;
  GEMM_CORE(A, Bt, (size_t)Q_, (size_t)Q_, Q_)

  float* Y = out + bz * (size_t)O_ * N_;
#pragma unroll
  for (int mi = 0; mi < 4; ++mi)
#pragma unroll
    for (int r = 0; r < 8; ++r) {
      size_t o = m0w + mi * 16 + r + hi * 8;
      float s = bn_scale[o] * rsqrtf(bn_var[o] + BN_EPS_);
      float bb = bn_bias[o] - bn_mean[o] * s;
#pragma unroll
      for (int ni = 0; ni < 4; ++ni) {
        size_t n = n0w + ni * 16 + col;
        float y = fmaf(acc[mi][ni][r], s, bb);
        Y[o * (size_t)N_ + n] = fmaxf(y, 0.f);
      }
    }
}

// ---------------------------------------------------------------------------
// Host launch
// ---------------------------------------------------------------------------
extern "C" void kernel_launch(void* const* d_in, const int* in_sizes, int n_in,
                              void* d_out, int out_size, void* d_ws, size_t ws_size,
                              hipStream_t stream) {
  const float* x        = (const float*)d_in[0];
  const float* query    = (const float*)d_in[1];
  const float* gamma    = (const float*)d_in[2];
  const float* conv_w   = (const float*)d_in[3];
  const float* bn_scale = (const float*)d_in[4];
  const float* bn_bias  = (const float*)d_in[5];
  const float* bn_mean  = (const float*)d_in[6];
  const float* bn_var   = (const float*)d_in[7];
  float* out = (float*)d_out;

  // Workspace layout (all offsets multiples of 256 B)
  constexpr size_t SZ_XB = (size_t)B_ * C_ * N_ * 2; // x bf16 row-major
  constexpr size_t SZ_XT = SZ_XB;                    // x bf16 transposed [N][C]
  constexpr size_t SZ_QB = (size_t)B_ * Q_ * N_ * 2; // query bf16
  constexpr size_t SZ_WB = (size_t)O_ * Q_ * 2;      // conv_w bf16
  constexpr size_t SZ_EN = (size_t)B_ * Q_ * C_ * 4; // energy fp32
  constexpr size_t SZ_AT = (size_t)B_ * Q_ * C_ * 2; // attention bf16
  char* ws = (char*)d_ws;
  __bf16* xb    = (__bf16*)(ws);
  __bf16* xTb   = (__bf16*)(ws + SZ_XB);
  __bf16* qb    = (__bf16*)(ws + SZ_XB + SZ_XT);
  __bf16* wb    = (__bf16*)(ws + SZ_XB + SZ_XT + SZ_QB);
  float*  energy= (float*) (ws + SZ_XB + SZ_XT + SZ_QB + SZ_WB);
  __bf16* att   = (__bf16*)(ws + SZ_XB + SZ_XT + SZ_QB + SZ_WB + SZ_EN);
  __bf16* outcT = (__bf16*)(ws + SZ_XB + SZ_XT + SZ_QB + SZ_WB + SZ_EN + SZ_AT);

  // 1) bf16 staging copies
  cvt_f32_to_bf16<<<4096, 256, 0, stream>>>(x, xb, (size_t)B_ * C_ * N_ / 4);
  cvt_f32_to_bf16<<<2048, 256, 0, stream>>>(query, qb, (size_t)B_ * Q_ * N_ / 4);
  cvt_f32_to_bf16<<<256, 256, 0, stream>>>(conv_w, wb, (size_t)O_ * Q_ / 4);
  transpose_cvt<<<dim3(N_ / 32, C_ / 32, B_), dim3(32, 8), 0, stream>>>(x, xTb);

  // 2) energy = Qf . Xf^T  [512 x 2560] per batch
  gemm_energy<<<dim3(C_ / 128, Q_ / 128, B_), 128, 0, stream>>>(qb, xb, energy);

  // 3) inverted-max softmax over C
  softmax_rows<<<B_ * Q_, 256, 0, stream>>>(energy, att);

  // 4) out_c = att . Xf, gamma*out_c + query, store transposed bf16
  gemm_attn<<<dim3(N_ / 128, Q_ / 128, B_), 128, 0, stream>>>(att, xTb, query, gamma, outcT);

  // 5) y = W . out_c, BN + ReLU
  gemm_conv_bn<<<dim3(N_ / 128, O_ / 128, B_), 128, 0, stream>>>(
      wb, outcT, bn_scale, bn_bias, bn_mean, bn_var, out);
}
